// CrossViewIPAttnProcessor_81338090652121
// MI455X (gfx1250) — matrix operand order repaired and checked by reading, not compiled, not run on hardware
//
#include <hip/hip_runtime.h>

typedef __attribute__((ext_vector_type(16))) _Float16 v16h;
typedef __attribute__((ext_vector_type(8)))  float    v8f;

#define HEADS 16
#define HDIM  64
constexpr int B_    = 8;
constexpr int LQ    = 4096;
constexpr int D_    = 1024;
constexpr int DC    = 768;
constexpr int LTEXT = 77;
constexpr int LIP   = 4;
constexpr int KPAD  = 80;   // text keys padded to 5 tiles of 16 (score N dim)
constexpr int KPAD2 = 96;   // padded to multiple of 32 (PV K dim)
constexpr int IPPAD = 16;
constexpr int IPPAD2= 32;
constexpr float ATT_SCALE = 0.125f;  // 1/sqrt(64)
constexpr float LN_EPSF   = 1e-5f;
constexpr float IP_SCALEF = 1.0f;

// ---- WMMA lane-layout helpers (CDNA5 ISA 7.12.2, wave32) ----
// 16-bit A 16x32: lane m=lane&15; this code's contiguous k order (B matches): element e -> k = e + (lane&16 ? 16:0)
__device__ __forceinline__ int akpat(int e, int hi) { return e + hi; }
// 16-bit B 32x16: lane n=lane&15; this code's contiguous k order (A matches): element e -> k = e + (lane&16 ? 16:0)
// f32 C/D 16x16: VGPR r -> row = r + (lane&16 ? 8:0); col = lane&15

__global__ void cvt_f32_to_f16(const float* __restrict__ src, _Float16* __restrict__ dst, int n) {
  int i = blockIdx.x * blockDim.x + threadIdx.x;
  int stride = gridDim.x * blockDim.x;
  for (; i < n; i += stride) dst[i] = (_Float16)src[i];
}

// LayerNorm text rows (pad to 80, zero-filled) + copy ip rows (pad to 16) as f16
__global__ void ln_prep(const float* __restrict__ enc, const float* __restrict__ lnw,
                        const float* __restrict__ lnb, _Float16* __restrict__ textln,
                        _Float16* __restrict__ ip16) {
  int b = blockIdx.y, r = blockIdx.x, t = threadIdx.x;
  __shared__ float red[256];
  if (r >= KPAD) {  // ip rows 80..95 -> ip16 rows 0..15
    int pr = r - KPAD;
    _Float16* dst = ip16 + ((size_t)b * IPPAD + pr) * DC;
    const float* src = enc + ((size_t)b * (LTEXT + LIP) + LTEXT + pr) * DC;
    for (int c = t; c < DC; c += 256) dst[c] = (pr < LIP) ? (_Float16)src[c] : (_Float16)0.f;
    return;
  }
  _Float16* dst = textln + ((size_t)b * KPAD + r) * DC;
  if (r >= LTEXT) { for (int c = t; c < DC; c += 256) dst[c] = (_Float16)0.f; return; }
  const float* src = enc + ((size_t)b * (LTEXT + LIP) + r) * DC;
  float x0 = src[t], x1 = src[t + 256], x2 = src[t + 512];
  red[t] = x0 + x1 + x2; __syncthreads();
  for (int o = 128; o > 0; o >>= 1) { if (t < o) red[t] += red[t + o]; __syncthreads(); }
  float mu = red[0] * (1.0f / DC); __syncthreads();
  float d0 = x0 - mu, d1 = x1 - mu, d2 = x2 - mu;
  red[t] = d0 * d0 + d1 * d1 + d2 * d2; __syncthreads();
  for (int o = 128; o > 0; o >>= 1) { if (t < o) red[t] += red[t + o]; __syncthreads(); }
  float rstd = rsqrtf(red[0] * (1.0f / DC) + LN_EPSF);
  dst[t]       = (_Float16)(d0 * rstd * lnw[t]       + lnb[t]);
  dst[t + 256] = (_Float16)(d1 * rstd * lnw[t + 256] + lnb[t + 256]);
  dst[t + 512] = (_Float16)(d2 * rstd * lnw[t + 512] + lnb[t + 512]);
}

// K/V (+ip) projections: out[m][n] = sum_k A[m][k] * W[n][k]   (WMMA f16, fp32 accum)
__global__ void kv_proj(const _Float16* __restrict__ textln, const _Float16* __restrict__ ip16,
                        const _Float16* __restrict__ Wk16, const _Float16* __restrict__ Wv16,
                        const _Float16* __restrict__ Wkip16, const _Float16* __restrict__ Wvip16,
                        _Float16* __restrict__ Kbuf, _Float16* __restrict__ Vt,
                        _Float16* __restrict__ ipK, _Float16* __restrict__ ipVt) {
  int b = blockIdx.x, ntile = blockIdx.y, z = blockIdx.z;   // z: 0 = K path, 1 = V path
  int wave = threadIdx.x >> 5, lane = threadIdx.x & 31;
  int nlo = lane & 15, hi = lane & 16;
  const v8f vzero = {};
  for (int t = wave; t < 6; t += 2) {   // 5 text m-tiles + 1 ip m-tile
    bool isip = (t == 5);
    const _Float16* A = isip ? (ip16 + (size_t)b * IPPAD * DC)
                             : (textln + ((size_t)b * KPAD + t * 16) * DC);
    const _Float16* W = (z == 0) ? (isip ? Wkip16 : Wk16) : (isip ? Wvip16 : Wv16);
    v8f acc = vzero;
    for (int ks = 0; ks < DC / 32; ++ks) {
      v16h a, bb;
#pragma unroll
      for (int e = 0; e < 16; ++e) {
        a[e]  = A[(size_t)nlo * DC + ks * 32 + akpat(e, hi)];
        bb[e] = W[(size_t)(ntile * 16 + nlo) * DC + ks * 32 + e + hi];
      }
      acc = __builtin_amdgcn_wmma_f32_16x16x32_f16(false, a, false, bb, (short)0, acc, false, false);
    }
    int d = ntile * 16 + nlo, h = d >> 6, dh = d & 63;
    size_t bh = (size_t)b * HEADS + h;
#pragma unroll
    for (int r = 0; r < 8; ++r) {
      int row = r + (hi >> 1);
      _Float16 vv = (_Float16)acc[r];
      if (z == 0) {
        if (!isip) Kbuf[(bh * KPAD + t * 16 + row) * HDIM + dh] = vv;
        else       ipK[(bh * IPPAD + row) * HDIM + dh] = vv;
      } else {
        if (!isip) Vt[(bh * HDIM + dh) * KPAD2 + t * 16 + row] = vv;   // transposed
        else       ipVt[(bh * HDIM + dh) * IPPAD2 + row] = vv;
      }
    }
  }
}

// Fused: Q tile (16 rows) = hs @ Wq^T, then text attention (77 keys) + ip attention (4 keys)
__global__ void __launch_bounds__(256) qattn(const float* __restrict__ hs,
                                             const _Float16* __restrict__ Wq16,
                                             const _Float16* __restrict__ Kbuf,
                                             const _Float16* __restrict__ Vt,
                                             const _Float16* __restrict__ ipK,
                                             const _Float16* __restrict__ ipVt,
                                             _Float16* __restrict__ attn) {
  __shared__ _Float16 Qs[16 * 1024];      // 32 KB
  __shared__ _Float16 Ps[8][16 * 96];     // 24 KB (Ps[0] doubles as hs staging in phase 1)
  __shared__ _Float16 Pip[8][16 * 32];    // 8 KB   -> 64 KB total
  int b = blockIdx.y;
  int q0 = blockIdx.x * 16;
  int tid = threadIdx.x, wave = tid >> 5, lane = tid & 31;
  int nlo = lane & 15, hi = lane & 16;
  const v8f vzero = {};

  // ---------------- phase 1: Q[16][1024] = hs_tile @ Wq^T ----------------
  _Float16* Ast = &Ps[0][0];              // 16x32 f16 staging chunk
  v8f qacc[8];
#pragma unroll
  for (int t = 0; t < 8; ++t) qacc[t] = vzero;
  for (int ks = 0; ks < D_ / 32; ++ks) {
    int i = tid;
#pragma unroll
    for (int it = 0; it < 2; ++it, i += 256) {   // 512 elems, 2 per thread
      int r = i >> 5, c = i & 31;
      Ast[r * 32 + c] = (_Float16)hs[((size_t)b * LQ + q0 + r) * D_ + ks * 32 + c];
    }
    __syncthreads();
    v16h a;
#pragma unroll
    for (int e = 0; e < 16; ++e) a[e] = Ast[nlo * 32 + akpat(e, hi)];
#pragma unroll
    for (int t = 0; t < 8; ++t) {
      int nt = wave * 8 + t;
      v16h bb;
#pragma unroll
      for (int e = 0; e < 16; ++e)
        bb[e] = Wq16[(size_t)(nt * 16 + nlo) * D_ + ks * 32 + e + hi];
      qacc[t] = __builtin_amdgcn_wmma_f32_16x16x32_f16(false, a, false, bb, (short)0, qacc[t], false, false);
    }
    __syncthreads();
  }
#pragma unroll
  for (int t = 0; t < 8; ++t) {
    int nt = wave * 8 + t;
#pragma unroll
    for (int r = 0; r < 8; ++r)
      Qs[(r + (hi >> 1)) * 1024 + nt * 16 + nlo] = (_Float16)qacc[t][r];
  }
  __syncthreads();

  // ---------------- phase 2: attention, one head per wave-task ----------------
  for (int task = 0; task < 2; ++task) {
    int h = wave + task * 8;
    size_t bh = (size_t)b * HEADS + h;
    v16h aq[2];
#pragma unroll
    for (int ks = 0; ks < 2; ++ks)
#pragma unroll
      for (int e = 0; e < 16; ++e)
        aq[ks][e] = Qs[nlo * 1024 + h * 64 + ks * 32 + akpat(e, hi)];

    // text scores: 5 key-tiles of 16
    v8f sc[5];
#pragma unroll
    for (int nt = 0; nt < 5; ++nt) {
      sc[nt] = vzero;
#pragma unroll
      for (int ks = 0; ks < 2; ++ks) {
        v16h bk;
#pragma unroll
        for (int e = 0; e < 16; ++e)
          bk[e] = Kbuf[(bh * KPAD + nt * 16 + nlo) * HDIM + ks * 32 + e + hi];
        sc[nt] = __builtin_amdgcn_wmma_f32_16x16x32_f16(false, aq[ks], false, bk, (short)0, sc[nt], false, false);
      }
    }
#pragma unroll
    for (int nt = 0; nt < 5; ++nt) {
      bool valid = (nt * 16 + nlo) < LTEXT;
#pragma unroll
      for (int r = 0; r < 8; ++r)
        sc[nt][r] = valid ? sc[nt][r] * ATT_SCALE : -3.0e38f;
    }
    // softmax: each row lives across 16 lanes of one half-wave
    float sminv[8];
#pragma unroll
    for (int r = 0; r < 8; ++r) {
      float m = sc[0][r];
#pragma unroll
      for (int nt = 1; nt < 5; ++nt) m = fmaxf(m, sc[nt][r]);
#pragma unroll
      for (int o = 1; o < 16; o <<= 1) m = fmaxf(m, __shfl_xor(m, o, 16));
      float s = 0.f;
#pragma unroll
      for (int nt = 0; nt < 5; ++nt) { sc[nt][r] = __expf(sc[nt][r] - m); s += sc[nt][r]; }
#pragma unroll
      for (int o = 1; o < 16; o <<= 1) s += __shfl_xor(s, o, 16);
      sminv[r] = 1.f / s;
    }

    // ip scores: 1 key-tile (4 valid keys), separate softmax
    v8f sip = vzero;
#pragma unroll
    for (int ks = 0; ks < 2; ++ks) {
      v16h bk;
#pragma unroll
      for (int e = 0; e < 16; ++e)
        bk[e] = ipK[(bh * IPPAD + nlo) * HDIM + ks * 32 + e + hi];
      sip = __builtin_amdgcn_wmma_f32_16x16x32_f16(false, aq[ks], false, bk, (short)0, sip, false, false);
    }
    {
      bool valid = nlo < LIP;
#pragma unroll
      for (int r = 0; r < 8; ++r) {
        float v = valid ? sip[r] * ATT_SCALE : -3.0e38f;
        float m = v;
#pragma unroll
        for (int o = 1; o < 16; o <<= 1) m = fmaxf(m, __shfl_xor(m, o, 16));
        float ev = __expf(v - m);
        float s = ev;
#pragma unroll
        for (int o = 1; o < 16; o <<= 1) s += __shfl_xor(s, o, 16);
        sip[r] = ev / s * IP_SCALEF;       // IP_SCALE folded into probs
      }
    }

    // repack probabilities (C layout -> A layout) via LDS, zero-padded keys
    _Float16* P  = Ps[wave];
    _Float16* Pi = Pip[wave];
    for (int i = lane; i < 16 * 96; i += 32) P[i]  = (_Float16)0.f;
    for (int i = lane; i < 16 * 32; i += 32) Pi[i] = (_Float16)0.f;
#pragma unroll
    for (int nt = 0; nt < 5; ++nt)
#pragma unroll
      for (int r = 0; r < 8; ++r)
        P[(r + (hi >> 1)) * 96 + nt * 16 + nlo] = (_Float16)(sc[nt][r] * sminv[r]);
#pragma unroll
    for (int r = 0; r < 8; ++r)
      Pi[(r + (hi >> 1)) * 32 + nlo] = (_Float16)sip[r];

    // PV: text (96-key padded) then ip (32-key padded) accumulated together
    v8f o4[4];
#pragma unroll
    for (int nt = 0; nt < 4; ++nt) o4[nt] = vzero;
#pragma unroll
    for (int ks = 0; ks < 3; ++ks) {
      v16h ap;
#pragma unroll
      for (int e = 0; e < 16; ++e) ap[e] = P[nlo * 96 + ks * 32 + akpat(e, hi)];
#pragma unroll
      for (int nt = 0; nt < 4; ++nt) {
        v16h bv;
#pragma unroll
        for (int e = 0; e < 16; ++e)
          bv[e] = Vt[(bh * HDIM + nt * 16 + nlo) * KPAD2 + ks * 32 + e + hi];
        o4[nt] = __builtin_amdgcn_wmma_f32_16x16x32_f16(false, ap, false, bv, (short)0, o4[nt], false, false);
      }
    }
    {
      v16h ap;
#pragma unroll
      for (int e = 0; e < 16; ++e) ap[e] = Pi[nlo * 32 + akpat(e, hi)];
#pragma unroll
      for (int nt = 0; nt < 4; ++nt) {
        v16h bv;
#pragma unroll
        for (int e = 0; e < 16; ++e)
          bv[e] = ipVt[(bh * HDIM + nt * 16 + nlo) * IPPAD2 + e + hi];
        o4[nt] = __builtin_amdgcn_wmma_f32_16x16x32_f16(false, ap, false, bv, (short)0, o4[nt], false, false);
      }
    }
#pragma unroll
    for (int nt = 0; nt < 4; ++nt)
#pragma unroll
      for (int r = 0; r < 8; ++r)
        attn[((size_t)b * LQ + q0 + r + (hi >> 1)) * D_ + h * 64 + nt * 16 + nlo] = (_Float16)o4[nt][r];
  }
}

// out = attn_f16 @ Wo^T + bo (fp32 out)
// Register-tiled: each wave computes a 32x64 tile (2x4 grid of 16x16 WMMA tiles),
// software-pipelined fragment loads (k+1 fragments issued before k-step WMMAs).
__global__ void __launch_bounds__(256) out_proj(const _Float16* __restrict__ attn,
                                                const _Float16* __restrict__ Wo16,
                                                const float* __restrict__ bo,
                                                float* __restrict__ out) {
  int wave = threadIdx.x >> 5, lane = threadIdx.x & 31;
  int nlo = lane & 15, hi = lane & 16;
  int wm = wave & 3, wn = wave >> 2;                   // 4 waves along M, 2 along N
  size_t row0 = (size_t)blockIdx.x * 128 + wm * 32;    // wave: rows [row0, row0+32)
  int col0 = blockIdx.y * 128 + wn * 64;               // wave: cols [col0, col0+64)

  const v8f vzero = {};
  v8f acc[2][4];
#pragma unroll
  for (int mi = 0; mi < 2; ++mi)
#pragma unroll
    for (int ni = 0; ni < 4; ++ni) acc[mi][ni] = vzero;

  v16h a[2], bb[4], an[2], bn[4];
#pragma unroll
  for (int mi = 0; mi < 2; ++mi)
#pragma unroll
    for (int e = 0; e < 16; ++e)
      a[mi][e] = attn[(row0 + mi * 16 + nlo) * D_ + akpat(e, hi)];
#pragma unroll
  for (int ni = 0; ni < 4; ++ni)
#pragma unroll
    for (int e = 0; e < 16; ++e)
      bb[ni][e] = Wo16[(size_t)(col0 + ni * 16 + nlo) * D_ + e + hi];

  for (int ks = 0; ks < D_ / 32; ++ks) {
    if (ks + 1 < D_ / 32) {   // prefetch next k-step fragments before this step's WMMAs
      int k1 = (ks + 1) * 32;
#pragma unroll
      for (int mi = 0; mi < 2; ++mi)
#pragma unroll
        for (int e = 0; e < 16; ++e)
          an[mi][e] = attn[(row0 + mi * 16 + nlo) * D_ + k1 + akpat(e, hi)];
#pragma unroll
      for (int ni = 0; ni < 4; ++ni)
#pragma unroll
        for (int e = 0; e < 16; ++e)
          bn[ni][e] = Wo16[(size_t)(col0 + ni * 16 + nlo) * D_ + k1 + e + hi];
    }
#pragma unroll
    for (int mi = 0; mi < 2; ++mi)
#pragma unroll
      for (int ni = 0; ni < 4; ++ni)
        acc[mi][ni] = __builtin_amdgcn_wmma_f32_16x16x32_f16(false, a[mi], false, bb[ni],
                                                             (short)0, acc[mi][ni], false, false);
#pragma unroll
    for (int mi = 0; mi < 2; ++mi) a[mi] = an[mi];
#pragma unroll
    for (int ni = 0; ni < 4; ++ni) bb[ni] = bn[ni];
  }

#pragma unroll
  for (int ni = 0; ni < 4; ++ni) {
    float bias = bo[col0 + ni * 16 + nlo];
#pragma unroll
    for (int mi = 0; mi < 2; ++mi)
#pragma unroll
      for (int r = 0; r < 8; ++r)
        out[(row0 + mi * 16 + r + (hi >> 1)) * D_ + col0 + ni * 16 + nlo] = acc[mi][ni][r] + bias;
  }
}

extern "C" void kernel_launch(void* const* d_in, const int* in_sizes, int n_in,
                              void* d_out, int out_size, void* d_ws, size_t ws_size,
                              hipStream_t stream) {
  (void)in_sizes; (void)n_in; (void)out_size; (void)ws_size;
  const float* hs   = (const float*)d_in[0];
  const float* enc  = (const float*)d_in[1];
  const float* Wq   = (const float*)d_in[2];
  const float* Wk   = (const float*)d_in[3];
  const float* Wv   = (const float*)d_in[4];
  const float* Wkip = (const float*)d_in[5];
  const float* Wvip = (const float*)d_in[6];
  const float* Wo   = (const float*)d_in[7];
  const float* bo   = (const float*)d_in[8];
  const float* lnw  = (const float*)d_in[9];
  const float* lnb  = (const float*)d_in[10];
  float* out = (float*)d_out;

  size_t off = 0;
  auto carve = [&](size_t bytes) -> void* {
    void* p = (char*)d_ws + off;
    off += (bytes + 255) & ~(size_t)255;
    return p;
  };
  _Float16* Wq16   = (_Float16*)carve((size_t)1024 * 1024 * 2);
  _Float16* Wo16   = (_Float16*)carve((size_t)1024 * 1024 * 2);
  _Float16* Wk16   = (_Float16*)carve((size_t)1024 * 768 * 2);
  _Float16* Wv16   = (_Float16*)carve((size_t)1024 * 768 * 2);
  _Float16* Wkip16 = (_Float16*)carve((size_t)1024 * 768 * 2);
  _Float16* Wvip16 = (_Float16*)carve((size_t)1024 * 768 * 2);
  _Float16* textln = (_Float16*)carve((size_t)B_ * KPAD * DC * 2);
  _Float16* ip16   = (_Float16*)carve((size_t)B_ * IPPAD * DC * 2);
  _Float16* Kbuf   = (_Float16*)carve((size_t)B_ * HEADS * KPAD * HDIM * 2);
  _Float16* Vt     = (_Float16*)carve((size_t)B_ * HEADS * HDIM * KPAD2 * 2);
  _Float16* ipK    = (_Float16*)carve((size_t)B_ * HEADS * IPPAD * HDIM * 2);
  _Float16* ipVt   = (_Float16*)carve((size_t)B_ * HEADS * HDIM * IPPAD2 * 2);
  _Float16* attn   = (_Float16*)carve((size_t)B_ * LQ * D_ * 2);

  // zero padded key columns of Vt/ipVt (keys >= 80 / >= 16 never written by kv_proj)
  hipMemsetAsync(Vt,   0, (size_t)B_ * HEADS * HDIM * KPAD2 * 2, stream);
  hipMemsetAsync(ipVt, 0, (size_t)B_ * HEADS * HDIM * IPPAD2 * 2, stream);

  cvt_f32_to_f16<<<512, 256, 0, stream>>>(Wq,   Wq16,   1024 * 1024);
  cvt_f32_to_f16<<<512, 256, 0, stream>>>(Wo,   Wo16,   1024 * 1024);
  cvt_f32_to_f16<<<512, 256, 0, stream>>>(Wk,   Wk16,   1024 * 768);
  cvt_f32_to_f16<<<512, 256, 0, stream>>>(Wv,   Wv16,   1024 * 768);
  cvt_f32_to_f16<<<512, 256, 0, stream>>>(Wkip, Wkip16, 1024 * 768);
  cvt_f32_to_f16<<<512, 256, 0, stream>>>(Wvip, Wvip16, 1024 * 768);

  ln_prep<<<dim3(KPAD + IPPAD, B_), 256, 0, stream>>>(enc, lnw, lnb, textln, ip16);
  kv_proj<<<dim3(B_, D_ / 16, 2), 64, 0, stream>>>(textln, ip16, Wk16, Wv16, Wkip16, Wvip16,
                                                   Kbuf, Vt, ipK, ipVt);
  qattn<<<dim3(LQ / 16, B_), 256, 0, stream>>>(hs, Wq16, Kbuf, Vt, ipK, ipVt, attn);
  out_proj<<<dim3((B_ * LQ) / 128, D_ / 128), 256, 0, stream>>>(attn, Wo16, bo, out);
}
